// KeyFeatureExtractor_29059748725387
// MI455X (gfx1250) — matrix operand order repaired and checked by reading, not compiled, not run on hardware
//
#include <hip/hip_runtime.h>
#include <hip/hip_bf16.h>
#include <stdint.h>

// ---------------- problem constants ----------------
#define BB 8
#define TT 2048
#define CC 1024
#define NH 16
#define HS 64
#define LCH 128              // recurrence chunk length
#define NCH (TT / LCH)       // 16 chunks
#define EXP_SCALING_F 10.0f
#define KEY_SCALE_MAX_F 11.090354888959125f  // ln(2^16 - 1)

// ---------------- types ----------------
typedef __attribute__((ext_vector_type(16))) __bf16 v16bf;
typedef __attribute__((ext_vector_type(8)))  __bf16 bf16x8;
typedef __attribute__((ext_vector_type(4)))  __bf16 bf16x4;
typedef __attribute__((ext_vector_type(8)))  float  v8f;
typedef __attribute__((ext_vector_type(4)))  float  f32x4;

// ---------------- kernel 0: fp32 -> bf16 convert (vectorized) ----------------
__global__ __launch_bounds__(256) void cvt_f32_to_bf16(const float* __restrict__ in,
                                                       __bf16* __restrict__ out,
                                                       int n4) {
  int i = blockIdx.x * blockDim.x + threadIdx.x;
  if (i < n4) {
    f32x4 v = ((const f32x4*)in)[i];
    bf16x4 o;
    o[0] = (__bf16)v[0]; o[1] = (__bf16)v[1];
    o[2] = (__bf16)v[2]; o[3] = (__bf16)v[3];
    ((bf16x4*)out)[i] = o;
  }
}

// ---------------- kernel 1: bf16 WMMA GEMM  C(M,N) = A(M,K) * W(N,K)^T --------
#define BM 128
#define BN 64
#define BK 32
#define LDA 40                    // LDS row pitch in elements (32 data + 8 pad -> 80B)
#define KTILES (CC / BK)

__device__ __forceinline__ void async_b128(const void* gptr, void* lptr) {
  unsigned lds = (unsigned)(uintptr_t)lptr;                // low 32 bits = LDS offset
  unsigned long long g = (unsigned long long)(uintptr_t)gptr;
  asm volatile("global_load_async_to_lds_b128 %0, %1, off"
               :: "v"(lds), "v"(g) : "memory");
}

__global__ __launch_bounds__(256) void wmma_gemm_bf16(const __bf16* __restrict__ A,
                                                      const __bf16* __restrict__ W,
                                                      float* __restrict__ C) {
  __shared__ __bf16 As[2][BM * LDA];   // 20480 B
  __shared__ __bf16 Bs[2][BN * LDA];   // 10240 B

  const int tid  = threadIdx.x;
  const int lane = tid & 31;
  const int wid  = tid >> 5;        // 0..7
  const int wm   = wid >> 1;        // 0..3  (M sub-tile)
  const int wn   = wid & 1;         // 0..1  (N sub-tile)
  const int half = lane >> 4;       // 0/1
  const int l15  = lane & 15;

  const int gM0 = blockIdx.y * BM;
  const int gN0 = blockIdx.x * BN;

  // ---- async tile fill: A = 512 x 16B segs (2/thread), B = 256 segs (1/thread)
  auto loadTiles = [&](int buf, int k0) {
    {
      int s = tid;                      // first A segment
      int row = s >> 2, seg = s & 3;
      async_b128(A + (size_t)(gM0 + row) * CC + k0 + seg * 8,
                 &As[buf][row * LDA + seg * 8]);
      s = tid + 256;                    // second A segment
      row = s >> 2; seg = s & 3;
      async_b128(A + (size_t)(gM0 + row) * CC + k0 + seg * 8,
                 &As[buf][row * LDA + seg * 8]);
    }
    {
      int row = tid >> 2, seg = tid & 3;
      async_b128(W + (size_t)(gN0 + row) * CC + k0 + seg * 8,
                 &Bs[buf][row * LDA + seg * 8]);
    }
  };

  v8f acc[2][2] = {};

  loadTiles(0, 0);
  asm volatile("s_wait_asynccnt 0" ::: "memory");
  __syncthreads();

  for (int kt = 0; kt < KTILES; ++kt) {
    const int buf = kt & 1;
    if (kt + 1 < KTILES) loadTiles(buf ^ 1, (kt + 1) * BK);

    const __bf16* as = &As[buf][0];
    const __bf16* bs = &Bs[buf][0];

    v16bf af[2], bfx[2];
#pragma unroll
    for (int tm = 0; tm < 2; ++tm) {
      // A 16x32 bf16 layout: lane half 0 -> K{0..7,16..23}, half 1 -> K{8..15,24..31}
      int row = wm * 32 + tm * 16 + l15;
      bf16x8 lo = *(const bf16x8*)(as + row * LDA + half * 8);
      bf16x8 hi = *(const bf16x8*)(as + row * LDA + half * 8 + 16);
      af[tm] = __builtin_shufflevector(lo, hi, 0,1,2,3,4,5,6,7,8,9,10,11,12,13,14,15);
    }
#pragma unroll
    for (int tn = 0; tn < 2; ++tn) {
      // B 32x16 bf16 layout: N = l15, lane half 0 -> K{0..7,16..23}, half 1 -> K{8..15,24..31}
      int row = wn * 32 + tn * 16 + l15;
      bf16x8 lo = *(const bf16x8*)(bs + row * LDA + half * 8);
      bf16x8 hi = *(const bf16x8*)(bs + row * LDA + half * 8 + 16);
      bfx[tn] = __builtin_shufflevector(lo, hi, 0,1,2,3,4,5,6,7,8,9,10,11,12,13,14,15);
    }
#pragma unroll
    for (int tm = 0; tm < 2; ++tm)
#pragma unroll
      for (int tn = 0; tn < 2; ++tn)
        acc[tm][tn] = __builtin_amdgcn_wmma_f32_16x16x32_bf16(
            false, af[tm], false, bfx[tn], (short)0, acc[tm][tn], false, false);

    asm volatile("s_wait_asynccnt 0" ::: "memory");
    __syncthreads();
  }

  // C 16x16 f32 layout: VGPR r holds M = r + 8*half, N = l15
#pragma unroll
  for (int tm = 0; tm < 2; ++tm)
#pragma unroll
    for (int tn = 0; tn < 2; ++tn) {
      int col = gN0 + wn * 32 + tn * 16 + l15;
#pragma unroll
      for (int r = 0; r < 8; ++r) {
        int rowg = gM0 + wm * 32 + tm * 16 + half * 8 + r;
        C[(size_t)rowg * CC + col] = acc[tm][tn][r];
      }
    }
}

// ---------------- kernel 2: per-chunk local recurrence (phase A) --------------
// one wave per (b,h,chunk); lane owns d=lane and d=lane+32
__global__ __launch_bounds__(128) void leaky_local(const float* __restrict__ kproj,
                                                   const float* __restrict__ lkb,
                                                   float* __restrict__ yloc,
                                                   float* __restrict__ states) {
  int w = blockIdx.x * 4 + (threadIdx.x >> 5);   // 0..2047
  int lane = threadIdx.x & 31;
  int c = w % NCH;
  int bh = w / NCH;
  int h = bh % NH;
  int b = bh / NH;
  float rate = fabsf(lkb[h]) * EXP_SCALING_F;
  float a = __expf(-rate);
  float y0 = 0.f, y1 = 0.f;
  size_t inBase  = ((size_t)b * TT) * CC + h * HS + lane;
  size_t outBase = ((size_t)bh * TT) * HS + lane;
  int t0 = c * LCH;
  for (int tl = 0; tl < LCH; ++tl) {
    size_t ti = (size_t)(t0 + tl);
    float k0 = kproj[inBase + ti * CC];
    float k1 = kproj[inBase + ti * CC + 32];
    y0 = fmaf(a, y0, k0);
    y1 = fmaf(a, y1, k1);
    yloc[outBase + ti * HS]      = y0;
    yloc[outBase + ti * HS + 32] = y1;
  }
  size_t sb = ((size_t)bh * NCH + c) * HS + lane;
  states[sb] = y0;
  states[sb + 32] = y1;
}

// ---------------- kernel 3: scan chunk carries (phase B, tiny) ----------------
__global__ __launch_bounds__(128) void leaky_scan(const float* __restrict__ states,
                                                  const float* __restrict__ lkb,
                                                  float* __restrict__ carryin) {
  int w = blockIdx.x * 4 + (threadIdx.x >> 5);   // bh: 0..127
  int lane = threadIdx.x & 31;
  int h = w % NH;
  float rate = fabsf(lkb[h]) * EXP_SCALING_F;
  float aLC = __expf(-rate * (float)LCH);
  float c0 = 0.f, c1 = 0.f;
  size_t base = (size_t)w * NCH * HS + lane;
  for (int c = 0; c < NCH; ++c) {
    carryin[base + c * HS]      = c0;
    carryin[base + c * HS + 32] = c1;
    c0 = fmaf(aLC, c0, states[base + c * HS]);
    c1 = fmaf(aLC, c1, states[base + c * HS + 32]);
  }
}

// ---------------- kernel 4: carry fixup + L2 norm + scale (phase C) -----------
// one wave per (b,h,t) row; intra-wave reduction, no barriers
__global__ __launch_bounds__(256) void fix_norm_scale(const float* __restrict__ yloc,
                                                      const float* __restrict__ carryin,
                                                      const float* __restrict__ lkb,
                                                      const float* __restrict__ ksc,
                                                      const int* __restrict__ spow,
                                                      float* __restrict__ out) {
  int w = blockIdx.x * 8 + (threadIdx.x >> 5);   // row id 0..262143
  int lane = threadIdx.x & 31;
  int t  = w % TT;
  int bh = w / TT;
  int h  = bh % NH;
  float rate = fabsf(lkb[h]) * EXP_SCALING_F;
  int c  = t / LCH;
  int tl = t - c * LCH;
  float af = __expf(-rate * (float)(tl + 1));
  size_t yb = (size_t)w * HS + lane;
  size_t cb = ((size_t)bh * NCH + c) * HS + lane;
  float y0 = fmaf(af, carryin[cb],      yloc[yb]);
  float y1 = fmaf(af, carryin[cb + 32], yloc[yb + 32]);
  float s = y0 * y0 + y1 * y1;
#pragma unroll
  for (int off = 16; off > 0; off >>= 1) s += __shfl_xor(s, off, 32);
  float sc  = __expf(fminf((float)spow[0] * EXP_SCALING_F * ksc[h], KEY_SCALE_MAX_F));
  float inv = sc / (sqrtf(s) + 1e-10f);
  out[yb]      = y0 * inv;
  out[yb + 32] = y1 * inv;
}

// ---------------- host launcher ----------------
extern "C" void kernel_launch(void* const* d_in, const int* in_sizes, int n_in,
                              void* d_out, int out_size, void* d_ws, size_t ws_size,
                              hipStream_t stream) {
  const float* x   = (const float*)d_in[0];   // (B,T,C)
  const float* Wk  = (const float*)d_in[1];   // (C,C)
  const float* lkb = (const float*)d_in[2];   // (nh)
  const float* ksc = (const float*)d_in[3];   // (nh)
  const int*  spow = (const int*)d_in[4];     // scalar

  const size_t M = (size_t)BB * TT;           // 16384
  const size_t nX = M * CC;                   // 16,777,216
  const size_t nW = (size_t)CC * CC;          // 1,048,576

  // workspace layout (bytes)
  char* wsb = (char*)d_ws;
  __bf16* xb    = (__bf16*)(wsb);                                  // 33,554,432 B
  __bf16* wb    = (__bf16*)(wsb + 33554432);                       //  2,097,152 B
  float*  kproj = (float*)(wsb + 35651584);                        // 67,108,864 B
  float*  yloc  = (float*)(wsb + 102760448);                       // 67,108,864 B
  float*  states  = (float*)(wsb + 169869312);                     //    524,288 B
  float*  carryin = (float*)(wsb + 170393600);                     //    524,288 B

  // 0) convert x, W to bf16
  cvt_f32_to_bf16<<<dim3((unsigned)(nX / 4 / 256)), dim3(256), 0, stream>>>(x, xb, (int)(nX / 4));
  cvt_f32_to_bf16<<<dim3((unsigned)(nW / 4 / 256)), dim3(256), 0, stream>>>(Wk, wb, (int)(nW / 4));

  // 1) kproj = x @ Wk^T  (M=16384, N=1024, K=1024)
  wmma_gemm_bf16<<<dim3(CC / BN, (unsigned)(M / BM)), dim3(256), 0, stream>>>(xb, wb, kproj);

  // 2) local decay recurrence per chunk: 2048 waves
  leaky_local<<<dim3(512), dim3(128), 0, stream>>>(kproj, lkb, yloc, states);

  // 3) carry scan across chunks: 128 waves
  leaky_scan<<<dim3(32), dim3(128), 0, stream>>>(states, lkb, carryin);

  // 4) fixup + L2 normalize + per-head scale -> output (B,nh,T,hs)
  fix_norm_scale<<<dim3(32768), dim3(256), 0, stream>>>(yloc, carryin, lkb, ksc, spow,
                                                        (float*)d_out);
}
